// dgcn_block_36790689857950
// MI455X (gfx1250) — matrix operand order repaired and checked by reading, not compiled, not run on hardware
//
#include <hip/hip_runtime.h>
#include <hip/hip_bf16.h>
#include <math.h>

#define NODES   100000
#define EDGES   1600000
#define ETOT    (EDGES + NODES)   /* 1,700,000 edges incl. self-loops */
#define INC     64
#define HEADS   4
#define CPH     32
#define HC      128
#define GN_EPS  1e-5f
#define NSLOPE  0.2f
#define RED_ROWS 256

typedef __attribute__((ext_vector_type(16))) _Float16 v16h;
typedef __attribute__((ext_vector_type(8)))  float    v8f;

// ---- order-preserving float<->uint mapping for atomic max on signed floats ----
__device__ __forceinline__ unsigned ordf(float f) {
  unsigned u = __float_as_uint(f);
  return (u & 0x80000000u) ? ~u : (u | 0x80000000u);
}
__device__ __forceinline__ float unordf(unsigned v) {
  return (v & 0x80000000u) ? __uint_as_float(v ^ 0x80000000u) : __uint_as_float(~v);
}

// ---------------------------------------------------------------------------
// Init: pre-norm out = bias, seg-max = ord(-inf), denom = 0, stat accs = 0
// ---------------------------------------------------------------------------
__global__ void k_init(float* __restrict__ pre, const float* __restrict__ bias,
                       unsigned* __restrict__ mbuf, float* __restrict__ den,
                       float* __restrict__ macc, float* __restrict__ vacc) {
  int t = blockIdx.x * blockDim.x + threadIdx.x;
  if (t < NODES * HC)    pre[t] = bias[t & (HC - 1)];
  if (t < NODES * HEADS) { mbuf[t] = 0x007FFFFFu; /* ord(-inf) */ den[t] = 0.0f; }
  if (t < HC)            { macc[t] = 0.0f; vacc[t] = 0.0f; }
}

// ---------------------------------------------------------------------------
// Node transforms x_l = x@W_l, x_r = x@W_r via v_wmma_f32_16x16x32_f16.
// Block = 256 thr = 8 waves; wave w owns output cols [16w, 16w+16).
// M-tile = 16 rows (100000 % 16 == 0). K=64 -> 2 WMMA k-steps.
// ---------------------------------------------------------------------------
__global__ void k_xform(const float* __restrict__ x, const float* __restrict__ Wl,
                        const float* __restrict__ Wr, float* __restrict__ xl,
                        float* __restrict__ xr) {
  const int wave = threadIdx.x >> 5;   // N-tile 0..7
  const int lane = threadIdx.x & 31;
  const int half = lane >> 4;
  const int m    = lane & 15;          // A row / C column-lane index
  const int rowbase = blockIdx.x * 16;
  const int col = wave * 16 + m;       // output column for B/C

  v8f cl = {}; v8f cr = {};
#pragma unroll
  for (int ks = 0; ks < 2; ++ks) {
    v16h a, bl, br;
    const float* xrow = x + (size_t)(rowbase + m) * INC + ks * 32;
#pragma unroll
    for (int v = 0; v < 8; ++v) {
      // A 16x32 f16 layout: VGPR v<4 -> K = half*8 + 2v ; v>=4 -> K = 16 + half*8 + 2(v-4)
      int kb = (v < 4) ? (half * 8 + 2 * v) : (16 + half * 8 + 2 * (v - 4));
      a[2 * v]     = (_Float16)xrow[kb];
      a[2 * v + 1] = (_Float16)xrow[kb + 1];
      // B 32x16 f16 layout: same K order as A (VGPR v<4 -> K = half*8 + 2v ; v>=4 -> K = 16 + half*8 + 2(v-4))
      int k = ks * 32 + kb;
      bl[2 * v]     = (_Float16)Wl[(size_t)k * HC + col];
      bl[2 * v + 1] = (_Float16)Wl[(size_t)(k + 1) * HC + col];
      br[2 * v]     = (_Float16)Wr[(size_t)k * HC + col];
      br[2 * v + 1] = (_Float16)Wr[(size_t)(k + 1) * HC + col];
    }
    cl = __builtin_amdgcn_wmma_f32_16x16x32_f16(false, a, false, bl, (short)0, cl, false, false);
    cr = __builtin_amdgcn_wmma_f32_16x16x32_f16(false, a, false, br, (short)0, cr, false, false);
  }
#pragma unroll
  for (int i = 0; i < 8; ++i) {
    int row = rowbase + half * 8 + i;  // C/D layout: VGPR i = row i (lanes<16) / 8+i
    xl[(size_t)row * HC + col] = cl[i];
    xr[(size_t)row * HC + col] = cr[i];
  }
}

// ---------------------------------------------------------------------------
// Pass 1: per-(edge,head) attention logit + segment max over dst
// ---------------------------------------------------------------------------
__global__ void k_score(const float* __restrict__ xl, const float* __restrict__ xr,
                        const long long* __restrict__ ei, const float* __restrict__ att,
                        float* __restrict__ ebuf, unsigned* __restrict__ mbuf) {
  int t = blockIdx.x * blockDim.x + threadIdx.x;
  if (t >= ETOT * HEADS) return;
  int e = t >> 2, h = t & 3;
  long long s, d;
  if (e < EDGES) { s = ei[e]; d = ei[EDGES + e]; } else { s = d = (long long)(e - EDGES); }
  const float4* pl = (const float4*)(xl + (size_t)s * HC + h * CPH);
  const float4* pr = (const float4*)(xr + (size_t)d * HC + h * CPH);
  const float4* pa = (const float4*)(att + h * CPH);
  float acc = 0.f;
#pragma unroll
  for (int q = 0; q < 8; ++q) {
    float4 a = pl[q], b = pr[q], w = pa[q];
    float v;
    v = a.x + b.x; v = v > 0.f ? v : NSLOPE * v; acc += v * w.x;
    v = a.y + b.y; v = v > 0.f ? v : NSLOPE * v; acc += v * w.y;
    v = a.z + b.z; v = v > 0.f ? v : NSLOPE * v; acc += v * w.z;
    v = a.w + b.w; v = v > 0.f ? v : NSLOPE * v; acc += v * w.w;
  }
  ebuf[t] = acc;
  atomicMax(&mbuf[(size_t)d * HEADS + h], ordf(acc));
}

// ---------------------------------------------------------------------------
// Pass 2: ex = exp(e - max[dst]); segment sum of ex over dst
// ---------------------------------------------------------------------------
__global__ void k_exp(const long long* __restrict__ ei, float* __restrict__ ebuf,
                      const unsigned* __restrict__ mbuf, float* __restrict__ den) {
  int t = blockIdx.x * blockDim.x + threadIdx.x;
  if (t >= ETOT * HEADS) return;
  int e = t >> 2, h = t & 3;
  long long d = (e < EDGES) ? ei[EDGES + e] : (long long)(e - EDGES);
  float m = unordf(mbuf[(size_t)d * HEADS + h]);
  float ex = __expf(ebuf[t] - m);
  ebuf[t] = ex;
  atomicAdd(&den[(size_t)d * HEADS + h], ex);
}

// ---------------------------------------------------------------------------
// Pass 3: scatter alpha * x_l[src] into pre-norm output (bias pre-loaded)
// One thread per (edge, 4 channels): ETOT*32 threads.
// ---------------------------------------------------------------------------
__global__ void k_scatter(const float* __restrict__ xl, const long long* __restrict__ ei,
                          const float* __restrict__ ebuf, const float* __restrict__ den,
                          float* __restrict__ pre) {
  int t = blockIdx.x * blockDim.x + threadIdx.x;
  if (t >= ETOT * 32) return;
  int e = t >> 5, g = t & 31;      // g indexes groups of 4 channels
  int h = g >> 3;                  // head = channel / 32
  long long s, d;
  if (e < EDGES) { s = ei[e]; d = ei[EDGES + e]; } else { s = d = (long long)(e - EDGES); }
  float alpha = ebuf[(size_t)e * HEADS + h] / (den[(size_t)d * HEADS + h] + 1e-16f);
  float4 xv = *(const float4*)(xl + (size_t)s * HC + g * 4);
  float* o = pre + (size_t)d * HC + g * 4;
  atomicAdd(o + 0, xv.x * alpha);
  atomicAdd(o + 1, xv.y * alpha);
  atomicAdd(o + 2, xv.z * alpha);
  atomicAdd(o + 3, xv.w * alpha);
}

// ---------------------------------------------------------------------------
// GraphNorm reductions: column sums, then shifted^2 sums
// ---------------------------------------------------------------------------
__global__ void k_colsum(const float* __restrict__ pre, float* __restrict__ macc) {
  int j = threadIdx.x;  // 0..127
  size_t r0 = (size_t)blockIdx.x * RED_ROWS;
  size_t r1 = r0 + RED_ROWS; if (r1 > NODES) r1 = NODES;
  float s = 0.f;
  for (size_t r = r0; r < r1; ++r) s += pre[r * HC + j];
  atomicAdd(&macc[j], s);
}

__global__ void k_varsum(const float* __restrict__ pre, const float* __restrict__ macc,
                         const float* __restrict__ gms, float* __restrict__ vacc) {
  int j = threadIdx.x;
  float ms = gms[j] * (macc[j] * (1.0f / NODES));
  size_t r0 = (size_t)blockIdx.x * RED_ROWS;
  size_t r1 = r0 + RED_ROWS; if (r1 > NODES) r1 = NODES;
  float s = 0.f;
  for (size_t r = r0; r < r1; ++r) { float v = pre[r * HC + j] - ms; s += v * v; }
  atomicAdd(&vacc[j], s);
}

__global__ void k_final(const float* __restrict__ pre, const float* __restrict__ macc,
                        const float* __restrict__ vacc, const float* __restrict__ gms,
                        const float* __restrict__ gw, const float* __restrict__ gb,
                        float* __restrict__ out) {
  int t = blockIdx.x * blockDim.x + threadIdx.x;
  if (t >= NODES * HC) return;
  int j = t & (HC - 1);
  float mean = macc[j] * (1.0f / NODES);
  float sh   = pre[t] - gms[j] * mean;
  float var  = vacc[j] * (1.0f / NODES);
  out[t] = gw[j] * sh * rsqrtf(var + GN_EPS) + gb[j];
}

// ---------------------------------------------------------------------------
extern "C" void kernel_launch(void* const* d_in, const int* in_sizes, int n_in,
                              void* d_out, int out_size, void* d_ws, size_t ws_size,
                              hipStream_t stream) {
  const float*     x    = (const float*)d_in[0];
  const long long* ei   = (const long long*)d_in[1];   // int64 edge_index [2,E]
  const float*     Wl   = (const float*)d_in[2];
  const float*     Wr   = (const float*)d_in[3];
  const float*     att  = (const float*)d_in[4];
  const float*     bias = (const float*)d_in[5];
  const float*     gw   = (const float*)d_in[6];
  const float*     gb   = (const float*)d_in[7];
  const float*     gms  = (const float*)d_in[8];
  float*           out  = (float*)d_out;

  char* ws = (char*)d_ws;
  size_t off = 0;
  auto carve = [&](size_t bytes) -> char* {
    char* p = ws + off;
    off += (bytes + 255) & ~(size_t)255;
    return p;
  };
  float*    xl   = (float*)carve((size_t)NODES * HC * 4);
  float*    xr   = (float*)carve((size_t)NODES * HC * 4);
  float*    ebuf = (float*)carve((size_t)ETOT * HEADS * 4);
  unsigned* mbuf = (unsigned*)carve((size_t)NODES * HEADS * 4);
  float*    den  = (float*)carve((size_t)NODES * HEADS * 4);
  float*    pre  = (float*)carve((size_t)NODES * HC * 4);
  float*    macc = (float*)carve(HC * 4);
  float*    vacc = (float*)carve(HC * 4);

  const dim3 b256(256);
  const int eh = ETOT * HEADS;

  k_init   <<<(NODES * HC + 255) / 256, b256, 0, stream>>>(pre, bias, mbuf, den, macc, vacc);
  k_xform  <<<NODES / 16,               b256, 0, stream>>>(x, Wl, Wr, xl, xr);
  k_score  <<<(eh + 255) / 256,         b256, 0, stream>>>(xl, xr, ei, att, ebuf, mbuf);
  k_exp    <<<(eh + 255) / 256,         b256, 0, stream>>>(ei, ebuf, mbuf, den);
  k_scatter<<<(ETOT * 32 + 255) / 256,  b256, 0, stream>>>(xl, ei, ebuf, den, pre);
  k_colsum <<<(NODES + RED_ROWS - 1) / RED_ROWS, dim3(HC), 0, stream>>>(pre, macc);
  k_varsum <<<(NODES + RED_ROWS - 1) / RED_ROWS, dim3(HC), 0, stream>>>(pre, macc, gms, vacc);
  k_final  <<<(NODES * HC + 255) / 256, b256, 0, stream>>>(pre, macc, vacc, gms, gw, gb, out);
}
